// Block_18494129177041
// MI455X (gfx1250) — compile-verified, matrix operand order checked by reading, not run on hardware
//
#include <hip/hip_runtime.h>
#include <hip/hip_bf16.h>

// ---------------------------------------------------------------------------
// GPT block for MI455X (gfx1250, wave32, WMMA).
// ~138 GFLOP of matmul on an ~85MB working set (fits in 192MB L2) =>
// compute-bound on the matrix pipes. All matmuls run on
// v_wmma_f32_16x16x32_bf16 (f32 accum). GEMMs stage block tiles into LDS via
// the CDNA5 async Global->LDS path (ASYNCcnt) with double buffering so copy
// latency hides under the WMMA stream.
// ---------------------------------------------------------------------------

typedef __attribute__((ext_vector_type(16))) __bf16 v16bf;
typedef __attribute__((ext_vector_type(8)))  __bf16 v8bf;
typedef __attribute__((ext_vector_type(8)))  float  v8f;
typedef __attribute__((ext_vector_type(4)))  int    v4i;

#define N_EMBD 1024
#define N_HEADS 16
#define HEAD_DIM 64
#define SEQ_T 2048
#define BATCH 2
#define NTOK (BATCH * SEQ_T)   // 4096
#define C3 (3 * N_EMBD)        // 3072
#define CFC (4 * N_EMBD)       // 4096

#if defined(__AMDGCN__) && __has_builtin(__builtin_amdgcn_global_load_async_to_lds_b128)
#define HAVE_ASYNC_LDS 1
#endif

// 16-byte Global -> LDS copy. Async (ASYNCcnt-tracked, no VGPR staging) when
// the gfx1250 builtin is available; synchronous fallback otherwise.
// Builtin wants typed AS pointers: (v4i AS1* src, v4i AS3* dst, imm, imm cpol).
__device__ inline void copy16_g2l(const __bf16* g, __bf16* l) {
#ifdef HAVE_ASYNC_LDS
  __builtin_amdgcn_global_load_async_to_lds_b128(
      (__attribute__((address_space(1))) v4i*)g,
      (__attribute__((address_space(3))) v4i*)l, 0, 0);
#else
  *(v8bf*)l = *(const v8bf*)g;
#endif
}

__device__ inline void async_wait_all() {
#if defined(HAVE_ASYNC_LDS) && __has_builtin(__builtin_amdgcn_s_wait_asynccnt)
  __builtin_amdgcn_s_wait_asynccnt(0);
#endif
}

// Fragment loader for the CDNA5 16-bit A/B operand layout:
// lane l holds row (l&15); element j <-> k = kbase + (j>>3)*16 + (l>>4)*8 + (j&7)
// => two contiguous 16B loads at (rowbase + kbase + (l>>4)*8) and +16 elems.
// Works for both global and LDS pointers (ds_load_b128 for LDS).
__device__ inline v16bf load_frag(const __bf16* p) {
  v8bf lo = *(const v8bf*)(p);
  v8bf hi = *(const v8bf*)(p + 16);
  v16bf r;
#pragma unroll
  for (int i = 0; i < 8; ++i) { r[i] = lo[i]; r[i + 8] = hi[i]; }
  return r;
}

__device__ inline float gelu_f(float x) {
  const float c = 0.7978845608028654f;  // sqrt(2/pi)
  float t = tanhf(c * (x + 0.044715f * x * x * x));
  return 0.5f * x * (1.0f + t);
}

// ---------------------------------------------------------------------------
// Weight transpose + f32->bf16 cast.  W: Kd x Nd (row major) -> Wt: Nd x Kd.
// Makes every GEMM B-operand fragment a contiguous 2x16B load.
// ---------------------------------------------------------------------------
__global__ __launch_bounds__(256) void transpose_cast_kernel(
    const float* __restrict__ W, __bf16* __restrict__ Wt, int Kd, int Nd) {
  __shared__ float tile[32][33];
  const int n0 = blockIdx.x * 32, k0 = blockIdx.y * 32;
  const int tx = threadIdx.x & 31, ty = threadIdx.x >> 5;  // 32 x 8
#pragma unroll
  for (int j = 0; j < 4; ++j)
    tile[ty + j * 8][tx] = W[(size_t)(k0 + ty + j * 8) * Nd + n0 + tx];
  __syncthreads();
#pragma unroll
  for (int j = 0; j < 4; ++j)
    Wt[(size_t)(n0 + ty + j * 8) * Kd + k0 + tx] = (__bf16)tile[tx][ty + j * 8];
}

// ---------------------------------------------------------------------------
// LayerNorm (f32 in) -> bf16 out.  One row (1024) per block of 256 threads.
// ---------------------------------------------------------------------------
__global__ __launch_bounds__(256) void layernorm_bf16_kernel(
    const float* __restrict__ X, const float* __restrict__ w,
    const float* __restrict__ b, __bf16* __restrict__ out) {
  __shared__ float s1[256], s2[256];
  const int row = blockIdx.x;
  const float* x = X + (size_t)row * N_EMBD;
  float v[4], sum = 0.f, sq = 0.f;
#pragma unroll
  for (int i = 0; i < 4; ++i) {
    v[i] = x[threadIdx.x + 256 * i];
    sum += v[i];
    sq += v[i] * v[i];
  }
  s1[threadIdx.x] = sum;
  s2[threadIdx.x] = sq;
  __syncthreads();
  for (int off = 128; off > 0; off >>= 1) {
    if (threadIdx.x < off) {
      s1[threadIdx.x] += s1[threadIdx.x + off];
      s2[threadIdx.x] += s2[threadIdx.x + off];
    }
    __syncthreads();
  }
  const float mu = s1[0] * (1.0f / N_EMBD);
  const float var = s2[0] * (1.0f / N_EMBD) - mu * mu;
  const float rstd = rsqrtf(var + 1e-5f);
  __bf16* o = out + (size_t)row * N_EMBD;
#pragma unroll
  for (int i = 0; i < 4; ++i) {
    int c = threadIdx.x + 256 * i;
    o[c] = (__bf16)((v[i] - mu) * rstd * w[c] + b[c]);
  }
}

// ---------------------------------------------------------------------------
// WMMA GEMM:  C[M,N] = A[M,K] (bf16, row major) * Bt[N,K]^T (bf16, row major)
// Block = 256 threads = 8 waves; wave tile 32x64 (2x4 16x16 accums);
// block tile 64x256, k-step 32.
// Per k-step the block stages A (64x32, 4KB) and Bt (256x32, 16KB) slabs into
// LDS with async Global->LDS copies, double-buffered: stage s+1's copies are
// issued right after the stage-s barrier so they overlap the 64 WMMAs of
// stage s. Each thread owns 5 x 16B async copies per stage.
// MODE: 0 = bf16 out, 1 = +residual f32 out, 2 = GELU bf16.
// ---------------------------------------------------------------------------
template <int MODE>
__global__ __launch_bounds__(256) void gemm_bf16_wmma_kernel(
    const __bf16* __restrict__ A, const __bf16* __restrict__ Bt,
    void* __restrict__ outp, const float* __restrict__ Res, int M, int N,
    int K) {
  __shared__ __align__(16) __bf16 Abuf[2][64 * 32];    //  4KB x2
  __shared__ __align__(16) __bf16 Bbuf[2][256 * 32];   // 16KB x2
  const int tid = threadIdx.x;
  const int lane = tid & 31;
  const int wave = tid >> 5;
  const int hi8 = (lane >> 4) << 3;  // 0 or 8
  const int ln = lane & 15;
  const int mw = (wave >> 2) * 32;   // wave M offset inside block tile
  const int nw = (wave & 3) * 64;    // wave N offset inside block tile
  const int m0 = blockIdx.y * 64;
  const int n0 = blockIdx.x * 256;

  // --- per-thread copy slots: A = 1 chunk, B = 4 chunks of 16B ---
  const int arow = tid >> 2;               // 0..63
  const int ach = (tid & 3) * 8;           // element offset in k-slab
  const __bf16* agsrc = A + (size_t)(m0 + arow) * K + ach;
  const int aloff = arow * 32 + ach;
  const __bf16* bgsrc[4];
  int bloff[4];
#pragma unroll
  for (int i = 0; i < 4; ++i) {
    const int cid = tid + 256 * i;         // 0..1023
    const int brow = cid >> 2;             // 0..255
    const int bch = (cid & 3) * 8;
    bgsrc[i] = Bt + (size_t)(n0 + brow) * K + bch;
    bloff[i] = brow * 32 + bch;
  }

  v8f acc[2][4];
#pragma unroll
  for (int i = 0; i < 2; ++i)
#pragma unroll
    for (int j = 0; j < 4; ++j) acc[i][j] = (v8f){};

  // prologue: stage 0 copies
  copy16_g2l(agsrc, &Abuf[0][aloff]);
#pragma unroll
  for (int i = 0; i < 4; ++i) copy16_g2l(bgsrc[i], &Bbuf[0][bloff[i]]);

  const int nsteps = K >> 5;
  for (int s = 0; s < nsteps; ++s) {
    const int buf = s & 1;
    async_wait_all();      // my stage-s copies landed in LDS
    __syncthreads();       // everyone's did; all reads of buf^1 are finished
    if (s + 1 < nsteps) {  // overlap stage s+1 copies with stage-s WMMAs
      const int k = (s + 1) << 5;
      copy16_g2l(agsrc + k, &Abuf[buf ^ 1][aloff]);
#pragma unroll
      for (int i = 0; i < 4; ++i)
        copy16_g2l(bgsrc[i] + k, &Bbuf[buf ^ 1][bloff[i]]);
    }
    const v16bf a0 = load_frag(&Abuf[buf][(mw + ln) * 32 + hi8]);
    const v16bf a1 = load_frag(&Abuf[buf][(mw + 16 + ln) * 32 + hi8]);
#pragma unroll
    for (int j = 0; j < 4; ++j) {
      const v16bf bfr = load_frag(&Bbuf[buf][(nw + j * 16 + ln) * 32 + hi8]);
      acc[0][j] = __builtin_amdgcn_wmma_f32_16x16x32_bf16(
          false, a0, false, bfr, (short)0, acc[0][j], false, false);
      acc[1][j] = __builtin_amdgcn_wmma_f32_16x16x32_bf16(
          false, a1, false, bfr, (short)0, acc[1][j], false, false);
    }
    __syncthreads();  // all reads of buf done before stage s+2 overwrites it
  }

  // C layout: element (vgpr r, lane l) -> row r + 8*(l>>4), col (l&15)
#pragma unroll
  for (int mt = 0; mt < 2; ++mt)
#pragma unroll
    for (int nt = 0; nt < 4; ++nt)
#pragma unroll
      for (int r = 0; r < 8; ++r) {
        const int row = m0 + mw + mt * 16 + r + hi8;
        const int col = n0 + nw + nt * 16 + ln;
        const float v = acc[mt][nt][r];
        if (MODE == 0) {
          ((__bf16*)outp)[(size_t)row * N + col] = (__bf16)v;
        } else if (MODE == 1) {
          ((float*)outp)[(size_t)row * N + col] =
              v + Res[(size_t)row * N + col];
        } else {
          ((__bf16*)outp)[(size_t)row * N + col] = (__bf16)gelu_f(v);
        }
      }
}

// ---------------------------------------------------------------------------
// qkv [tok][3C] bf16 -> Q [bh][t][64], K [bh][t][64], Vt [bh][64][t]
// ---------------------------------------------------------------------------
__global__ __launch_bounds__(256) void permute_qkv_kernel(
    const __bf16* __restrict__ qkv, __bf16* __restrict__ Q,
    __bf16* __restrict__ Kk, __bf16* __restrict__ Vt) {
  const int idx = blockIdx.x * 256 + threadIdx.x;  // over NTOK * C
  const int c = idx & (N_EMBD - 1);
  const int tok = idx >> 10;
  const int b = tok >> 11;  // T = 2048
  const int t = tok & (SEQ_T - 1);
  const int h = c >> 6, d = c & 63;
  const int bh = b * N_HEADS + h;
  const __bf16* src = qkv + (size_t)tok * C3;
  Q[((size_t)bh * SEQ_T + t) * HEAD_DIM + d] = src[c];
  Kk[((size_t)bh * SEQ_T + t) * HEAD_DIM + d] = src[N_EMBD + c];
  Vt[((size_t)bh * HEAD_DIM + d) * SEQ_T + t] = src[2 * N_EMBD + c];
}

// ---------------------------------------------------------------------------
// Causal flash attention, fully on WMMA.
// Grid: (T/128, B*H); block 256 = 8 waves; wave owns 16 query rows.
// Per 32-key step: 4 score WMMAs + online softmax (half-wave shfl reductions
// match the 16x16 C-matrix lane layout) + P transposed through per-wave LDS
// into A-operand layout + 4 PV WMMAs (V stored [d][t] => contiguous B-frags).
// ---------------------------------------------------------------------------
__global__ __launch_bounds__(256) void attention_wmma_kernel(
    const __bf16* __restrict__ Q, const __bf16* __restrict__ Km,
    const __bf16* __restrict__ Vt, __bf16* __restrict__ Y) {
  __shared__ __align__(16) __bf16 Pbuf[8][16 * 32];  // per-wave P staging
  const int lane = threadIdx.x & 31;
  const int wave = threadIdx.x >> 5;
  const int hi8 = (lane >> 4) << 3;
  const int ln = lane & 15;
  const int bh = blockIdx.y;
  const int b = bh >> 4, h = bh & 15;
  const int q0 = blockIdx.x * 128 + wave * 16;

  const __bf16* Qh = Q + (size_t)bh * SEQ_T * HEAD_DIM;
  const __bf16* Kh = Km + (size_t)bh * SEQ_T * HEAD_DIM;
  const __bf16* Vh = Vt + (size_t)bh * HEAD_DIM * SEQ_T;

  const v16bf qa0 = load_frag(Qh + (size_t)(q0 + ln) * HEAD_DIM + hi8);
  const v16bf qa1 = load_frag(Qh + (size_t)(q0 + ln) * HEAD_DIM + 32 + hi8);

  v8f o[4];
#pragma unroll
  for (int t = 0; t < 4; ++t) o[t] = (v8f){};
  float mrow[8], lrow[8];
#pragma unroll
  for (int r = 0; r < 8; ++r) { mrow[r] = -1e30f; lrow[r] = 0.f; }

  __bf16* pb = &Pbuf[wave][0];

  for (int kb = 0; kb < q0 + 16; kb += 32) {
    // ---- scores S = (Q K^T) for 32 keys (two 16-key N tiles) ----
    v8f s[2];
#pragma unroll
    for (int g = 0; g < 2; ++g) {
      const __bf16* krow = Kh + (size_t)(kb + g * 16 + ln) * HEAD_DIM + hi8;
      v16bf kb_lo = load_frag(krow);        // k-dims 0..31
      v16bf kb_hi = load_frag(krow + 32);   // k-dims 32..63
      v8f z = (v8f){};
      z = __builtin_amdgcn_wmma_f32_16x16x32_bf16(false, qa0, false, kb_lo,
                                                  (short)0, z, false, false);
      s[g] = __builtin_amdgcn_wmma_f32_16x16x32_bf16(false, qa1, false, kb_hi,
                                                     (short)0, z, false, false);
    }
    // ---- online softmax; row r+8*(lane>>4) lives across one 16-lane half ----
#pragma unroll
    for (int r = 0; r < 8; ++r) {
      const int qg = q0 + r + hi8;
      float s0 = s[0][r] * 0.125f;  // 1/sqrt(64)
      float s1 = s[1][r] * 0.125f;
      s0 = (kb + ln <= qg) ? s0 : -3.0e38f;
      s1 = (kb + 16 + ln <= qg) ? s1 : -3.0e38f;
      float mx = fmaxf(s0, s1);
#pragma unroll
      for (int d = 1; d < 16; d <<= 1) mx = fmaxf(mx, __shfl_xor(mx, d, 32));
      const float mnew = fmaxf(mrow[r], mx);
      const float alpha = __expf(mrow[r] - mnew);
      mrow[r] = mnew;
      const float p0 = __expf(s0 - mnew);
      const float p1 = __expf(s1 - mnew);
      float rs = p0 + p1;
#pragma unroll
      for (int d = 1; d < 16; d <<= 1) rs += __shfl_xor(rs, d, 32);
      lrow[r] = lrow[r] * alpha + rs;
#pragma unroll
      for (int t = 0; t < 4; ++t) o[t][r] *= alpha;
      const int prow = r + hi8;
      pb[prow * 32 + ln] = (__bf16)p0;        // keys kb..kb+15
      pb[prow * 32 + 16 + ln] = (__bf16)p1;   // keys kb+16..kb+31
    }
    asm volatile("" ::: "memory");  // keep LDS stores before the re-read
    const v16bf pa = load_frag(pb + ln * 32 + hi8);  // P as A-operand 16x32
    // ---- O += P * V ;  V B-frags contiguous from Vt [dim][t] ----
#pragma unroll
    for (int t = 0; t < 4; ++t) {
      v16bf vb = load_frag(Vh + (size_t)(t * 16 + ln) * SEQ_T + kb + hi8);
      o[t] = __builtin_amdgcn_wmma_f32_16x16x32_bf16(false, pa, false, vb,
                                                     (short)0, o[t], false,
                                                     false);
    }
    asm volatile("" ::: "memory");
  }

  // ---- normalize and scatter back to [tok][C] bf16 ----
#pragma unroll
  for (int t = 0; t < 4; ++t)
#pragma unroll
    for (int r = 0; r < 8; ++r) {
      const int qg = q0 + r + hi8;
      const float v = o[t][r] / lrow[r];
      const int c = h * HEAD_DIM + t * 16 + ln;
      Y[((size_t)(b * SEQ_T + qg)) * N_EMBD + c] = (__bf16)v;
    }
}

// ---------------------------------------------------------------------------
// Orchestration
// ---------------------------------------------------------------------------
extern "C" void kernel_launch(void* const* d_in, const int* in_sizes, int n_in,
                              void* d_out, int out_size, void* d_ws,
                              size_t ws_size, hipStream_t stream) {
  const float* x = (const float*)d_in[0];
  const float* w_attn = (const float*)d_in[1];
  const float* w_proj = (const float*)d_in[2];
  const float* w_fc = (const float*)d_in[3];
  const float* w_fcp = (const float*)d_in[4];
  const float* ln1_w = (const float*)d_in[5];
  const float* ln1_b = (const float*)d_in[6];
  const float* ln2_w = (const float*)d_in[7];
  const float* ln2_b = (const float*)d_in[8];
  float* out = (float*)d_out;

  char* ws = (char*)d_ws;
  size_t off = 0;
  auto alloc = [&](size_t bytes) {
    void* p = ws + off;
    off = (off + bytes + 255) & ~(size_t)255;
    return p;
  };
  __bf16* wt_attn = (__bf16*)alloc((size_t)C3 * N_EMBD * 2);
  __bf16* wt_proj = (__bf16*)alloc((size_t)N_EMBD * N_EMBD * 2);
  __bf16* wt_fc = (__bf16*)alloc((size_t)CFC * N_EMBD * 2);
  __bf16* wt_fcp = (__bf16*)alloc((size_t)N_EMBD * CFC * 2);
  __bf16* ln1o = (__bf16*)alloc((size_t)NTOK * N_EMBD * 2);
  __bf16* qkv = (__bf16*)alloc((size_t)NTOK * C3 * 2);
  __bf16* Qb = (__bf16*)alloc((size_t)BATCH * N_HEADS * SEQ_T * HEAD_DIM * 2);
  __bf16* Kb = (__bf16*)alloc((size_t)BATCH * N_HEADS * SEQ_T * HEAD_DIM * 2);
  __bf16* Vtb = (__bf16*)alloc((size_t)BATCH * N_HEADS * HEAD_DIM * SEQ_T * 2);
  __bf16* yatt = (__bf16*)alloc((size_t)NTOK * N_EMBD * 2);
  float* x2 = (float*)alloc((size_t)NTOK * N_EMBD * 4);
  __bf16* ln2o = (__bf16*)alloc((size_t)NTOK * N_EMBD * 2);
  __bf16* hg = (__bf16*)alloc((size_t)NTOK * CFC * 2);

  // Weight prep: transpose + bf16 cast (weights fit in L2; cheap per call).
  transpose_cast_kernel<<<dim3(C3 / 32, N_EMBD / 32), 256, 0, stream>>>(
      w_attn, wt_attn, N_EMBD, C3);
  transpose_cast_kernel<<<dim3(N_EMBD / 32, N_EMBD / 32), 256, 0, stream>>>(
      w_proj, wt_proj, N_EMBD, N_EMBD);
  transpose_cast_kernel<<<dim3(CFC / 32, N_EMBD / 32), 256, 0, stream>>>(
      w_fc, wt_fc, N_EMBD, CFC);
  transpose_cast_kernel<<<dim3(N_EMBD / 32, CFC / 32), 256, 0, stream>>>(
      w_fcp, wt_fcp, CFC, N_EMBD);

  // 1) ln1
  layernorm_bf16_kernel<<<NTOK, 256, 0, stream>>>(x, ln1_w, ln1_b, ln1o);
  // 2) qkv = ln1 @ w_attn   (4096 x 3072 x 1024)
  gemm_bf16_wmma_kernel<0><<<dim3(C3 / 256, NTOK / 64), 256, 0, stream>>>(
      ln1o, wt_attn, qkv, nullptr, NTOK, C3, N_EMBD);
  // 3) split heads / transpose V
  permute_qkv_kernel<<<(NTOK * N_EMBD) / 256, 256, 0, stream>>>(qkv, Qb, Kb,
                                                                Vtb);
  // 4) attention
  attention_wmma_kernel<<<dim3(SEQ_T / 128, BATCH * N_HEADS), 256, 0,
                          stream>>>(Qb, Kb, Vtb, yatt);
  // 5) x2 = x + yatt @ w_proj
  gemm_bf16_wmma_kernel<1><<<dim3(N_EMBD / 256, NTOK / 64), 256, 0, stream>>>(
      yatt, wt_proj, x2, x, NTOK, N_EMBD, N_EMBD);
  // 6) ln2
  layernorm_bf16_kernel<<<NTOK, 256, 0, stream>>>(x2, ln2_w, ln2_b, ln2o);
  // 7) h = gelu(ln2 @ w_fc)  (4096 x 4096 x 1024)
  gemm_bf16_wmma_kernel<2><<<dim3(CFC / 256, NTOK / 64), 256, 0, stream>>>(
      ln2o, wt_fc, hg, nullptr, NTOK, CFC, N_EMBD);
  // 8) out = x2 + h @ w_fc_proj  (4096 x 1024 x 4096)
  gemm_bf16_wmma_kernel<1><<<dim3(N_EMBD / 256, NTOK / 64), 256, 0, stream>>>(
      hg, wt_fcp, out, x2, NTOK, N_EMBD, CFC);
}
